// HiPPO_RNN_3556232921441
// MI455X (gfx1250) — matrix operand order repaired and checked by reading, not compiled, not run on hardware
//
#include <hip/hip_runtime.h>
#include <hip/hip_bf16.h>
#include <cstdint>

typedef __attribute__((ext_vector_type(16))) __bf16 v16bf;
typedef __attribute__((ext_vector_type(8)))  __bf16 v8bf;
typedef __attribute__((ext_vector_type(4)))  __bf16 v4bf;
typedef __attribute__((ext_vector_type(8)))  float  v8f;

#define BM 128
#define BN 64
#define BK 32

#define HEADS 16
#define HD    64
#define BATCH 8
#define SEQ   2048

// Truncate a generic pointer to LDS to its 32-bit LDS byte address
// (flat LDS addresses carry the LDS offset in the low 32 bits).
__device__ __forceinline__ unsigned lds_addr32(const void* p) {
    return (unsigned)(uintptr_t)p;
}

// Per-lane async DMA: 16 bytes global -> LDS, tracked by ASYNCcnt.
__device__ __forceinline__ void async_b128(unsigned lds, unsigned long long gaddr) {
    asm volatile("global_load_async_to_lds_b128 %0, %1, off"
                 :: "v"(lds), "v"(gaddr) : "memory");
}

// ---------------------------------------------------------------------------
// fp32 -> bf16 elementwise (vector of 4 per thread)
// ---------------------------------------------------------------------------
__global__ __launch_bounds__(256) void cvt_f32_bf16(
    const float* __restrict__ in, __bf16* __restrict__ out, int n4)
{
    int i = blockIdx.x * 256 + threadIdx.x;
    if (i < n4) {
        float4 v = ((const float4*)in)[i];
        v4bf o;
        o[0] = (__bf16)v.x; o[1] = (__bf16)v.y;
        o[2] = (__bf16)v.z; o[3] = (__bf16)v.w;
        ((v4bf*)out)[i] = o;
    }
}

// ---------------------------------------------------------------------------
// Transpose + convert: in[K][N] fp32 -> out[N][K] bf16  (32x32 LDS tiles)
// ---------------------------------------------------------------------------
__global__ __launch_bounds__(256) void transpose_cvt(
    const float* __restrict__ in, __bf16* __restrict__ out, int Kdim, int Ndim)
{
    __shared__ float t[32][33];
    int x = blockIdx.x * 32 + threadIdx.x;   // n
    int y = blockIdx.y * 32 + threadIdx.y;   // k
#pragma unroll
    for (int j = 0; j < 32; j += 8)
        t[threadIdx.y + j][threadIdx.x] = in[(size_t)(y + j) * Ndim + x];
    __syncthreads();
    int xo = blockIdx.y * 32 + threadIdx.x;  // k
    int yo = blockIdx.x * 32 + threadIdx.y;  // n
#pragma unroll
    for (int j = 0; j < 32; j += 8)
        out[(size_t)(yo + j) * Kdim + xo] = (__bf16)t[threadIdx.x][threadIdx.y + j];
}

// ---------------------------------------------------------------------------
// bf16 WMMA GEMM with async global->LDS double buffering:
//   C[M,N] = A[M,K] @ Bt[N,K]^T (+ bias[N])
// A: bf16 row-major, Bt: bf16 pre-transposed [N][K].
// 256 threads = 8 waves; wave w computes rows [w*16, w*16+16) x 64 cols.
// Per K-step each thread issues 3x async B128 (A tile 8KB + B tile 4KB).
// ---------------------------------------------------------------------------
__global__ __launch_bounds__(256) void gemm_wmma_async(
    const __bf16* __restrict__ Ag, const __bf16* __restrict__ Bt,
    const float* __restrict__ bias, float* __restrict__ Cg,
    int M, int N, int K)
{
    __shared__ __bf16 Alds[2][BM * BK];   // [row][k] row-major, 8KB each
    __shared__ __bf16 Blds[2][BN * BK];   // [col][k], 4KB each

    const int tid  = threadIdx.x;
    const int lane = tid & 31;
    const int wave = tid >> 5;
    const int rowBase = blockIdx.y * BM;
    const int colBase = blockIdx.x * BN;

    // Byte-linear chunk assignment: LDS layout == (row*64 + within) bytes.
    const unsigned aB0 = tid * 16;           // A chunk 0 (first 4KB)
    const unsigned aB1 = 4096 + tid * 16;    // A chunk 1 (second 4KB)
    const unsigned bB  = tid * 16;           // B chunk (4KB)
    const int aR0 = aB0 >> 6, aO0 = aB0 & 63;
    const int aR1 = aB1 >> 6, aO1 = aB1 & 63;
    const int bR  = bB  >> 6, bO  = bB  & 63;

    const unsigned long long aBase = (unsigned long long)(uintptr_t)Ag;
    const unsigned long long bBase = (unsigned long long)(uintptr_t)Bt;

    auto issue_tile = [&](int kb, int buf) {
        unsigned la = lds_addr32(&Alds[buf][0]);
        unsigned lb = lds_addr32(&Blds[buf][0]);
        async_b128(la + aB0, aBase + ((size_t)(rowBase + aR0) * K + kb) * 2 + aO0);
        async_b128(la + aB1, aBase + ((size_t)(rowBase + aR1) * K + kb) * 2 + aO1);
        async_b128(lb + bB,  bBase + ((size_t)(colBase + bR ) * K + kb) * 2 + bO);
    };

    v8f acc[4];
#pragma unroll
    for (int t = 0; t < 4; ++t)
#pragma unroll
        for (int i = 0; i < 8; ++i) acc[t][i] = 0.0f;

    const int r   = lane & 15;
    const int khA = (lane >> 4) << 3;   // A frag: K {0..7,16..23} / {8..15,24..31}
    const int khB = (lane >> 4) << 3;   // B frag: K {0..7,16..23} / {8..15,24..31}

    issue_tile(0, 0);
    int buf = 0;
    for (int kb = 0; kb < K; kb += BK) {
        const bool hasNext = (kb + BK) < K;
        if (hasNext) {
            issue_tile(kb + BK, buf ^ 1);
            asm volatile("s_wait_asynccnt 0x3" ::: "memory");  // current tile landed
        } else {
            asm volatile("s_wait_asynccnt 0x0" ::: "memory");
        }
        __syncthreads();

        const __bf16* ap = &Alds[buf][(wave * 16 + r) * BK + khA];
        v8bf alo = *(const v8bf*)ap;
        v8bf ahi = *(const v8bf*)(ap + 16);
        v16bf afrag;
#pragma unroll
        for (int i = 0; i < 8; ++i) { afrag[i] = alo[i]; afrag[i + 8] = ahi[i]; }

#pragma unroll
        for (int t = 0; t < 4; ++t) {
            const __bf16* bp = &Blds[buf][(t * 16 + r) * BK + khB];
            v8bf blo = *(const v8bf*)bp;
            v8bf bhi = *(const v8bf*)(bp + 16);
            v16bf bfrag;
#pragma unroll
            for (int i = 0; i < 8; ++i) { bfrag[i] = blo[i]; bfrag[i + 8] = bhi[i]; }
            acc[t] = __builtin_amdgcn_wmma_f32_16x16x32_bf16(
                false, afrag, false, bfrag, (short)0, acc[t], false, false);
        }
        __syncthreads();   // all waves done reading buf before it is re-filled
        buf ^= 1;
    }

    const int mhi = (lane >> 4) * 8;
#pragma unroll
    for (int t = 0; t < 4; ++t) {
        int col = colBase + t * 16 + (lane & 15);
        float bv = bias ? bias[col] : 0.0f;
#pragma unroll
        for (int v = 0; v < 8; ++v) {
            int row = rowBase + wave * 16 + mhi + v;
            Cg[(size_t)row * N + col] = acc[t][v] + bv;
        }
    }
}

// ---------------------------------------------------------------------------
// Sequential HiPPO scan. One workgroup per head (grid=16), 128 threads
// (4 waves; wave w owns output columns [w*16, w*16+16)).
// h-update via WMMA bf16 (batch 8 padded into 16-row tile);
// c-update (numerically sensitive recurrence) in fp32 VALU.
// Emits states in bf16 for the downstream WMMA GEMM.
// ---------------------------------------------------------------------------
__global__ __launch_bounds__(128) void hippo_scan(
    const float* __restrict__ proj,     // [B][S][1024]
    const float* __restrict__ state_w,  // [16][64][64]
    const float* __restrict__ hippo_w,  // [16][64][64]
    const float* __restrict__ comp_w,   // [16][64]
    const float* __restrict__ Amat,     // [64][64]
    const float* __restrict__ Bvec,     // [64]
    __bf16* __restrict__ states_bf,     // [B][S][1024] bf16
    float* __restrict__ out_state)      // [B][1024]
{
    __shared__ __bf16 Wt[HD * HD];
    __shared__ __bf16 Ht[HD * HD];
    __shared__ float  Alds[HD * HD];
    __shared__ float  cw[HD];
    __shared__ float  bv[HD];
    __shared__ float  hbuf[BATCH * HD];
    __shared__ float  cbuf[2][BATCH * HD];
    __shared__ float  comp[BATCH];

    const int head = blockIdx.x;
    const int tid  = threadIdx.x;
    const int lane = tid & 31;
    const int wave = tid >> 5;

    const float* SW = state_w + head * HD * HD;
    const float* HW = hippo_w + head * HD * HD;

    for (int i = tid; i < HD * HD; i += 128) {
        int k = i >> 6;
        int n = i & 63;
        Wt[n * HD + k] = (__bf16)SW[i];
        Ht[n * HD + k] = (__bf16)HW[i];
        Alds[i] = Amat[i];
    }
    for (int i = tid; i < HD; i += 128) { cw[i] = comp_w[head * HD + i]; bv[i] = Bvec[i]; }
    for (int i = tid; i < BATCH * HD; i += 128) { hbuf[i] = 0.0f; cbuf[0][i] = 0.0f; }
    if (tid < BATCH) comp[tid] = 0.0f;
    int cur = 0;
    __syncthreads();

    const int r   = lane & 15;
    const int khA = (lane >> 4) << 3;
    const int khB = (lane >> 4) << 3;
    const int colsBase = head * HD + wave * 16;

    for (int s = 0; s < SEQ; ++s) {
        v8f acc;
#pragma unroll
        for (int i = 0; i < 8; ++i) acc[i] = 0.0f;

#pragma unroll
        for (int kb = 0; kb < 2; ++kb) {
            v16bf afrag;
            if (r < BATCH) {
                const float* hp = &hbuf[r * HD + kb * 32 + khA];
#pragma unroll
                for (int i = 0; i < 8; ++i) { afrag[i] = (__bf16)hp[i]; afrag[i + 8] = (__bf16)hp[i + 16]; }
            } else {
#pragma unroll
                for (int i = 0; i < 16; ++i) afrag[i] = (__bf16)0.0f;
            }
            const __bf16* bp = &Wt[(wave * 16 + r) * HD + kb * 32 + khB];
            v8bf blo = *(const v8bf*)bp;
            v8bf bhi = *(const v8bf*)(bp + 16);
            v16bf bfrag;
#pragma unroll
            for (int i = 0; i < 8; ++i) { bfrag[i] = blo[i]; bfrag[i + 8] = bhi[i]; }
            acc = __builtin_amdgcn_wmma_f32_16x16x32_bf16(
                false, afrag, false, bfrag, (short)0, acc, false, false);
        }
#pragma unroll
        for (int kb = 0; kb < 2; ++kb) {
            v16bf afrag;
            if (r < BATCH) {
                const float* cp = &cbuf[cur][r * HD + kb * 32 + khA];
#pragma unroll
                for (int i = 0; i < 8; ++i) { afrag[i] = (__bf16)cp[i]; afrag[i + 8] = (__bf16)cp[i + 16]; }
            } else {
#pragma unroll
                for (int i = 0; i < 16; ++i) afrag[i] = (__bf16)0.0f;
            }
            const __bf16* bp = &Ht[(wave * 16 + r) * HD + kb * 32 + khB];
            v8bf blo = *(const v8bf*)bp;
            v8bf bhi = *(const v8bf*)(bp + 16);
            v16bf bfrag;
#pragma unroll
            for (int i = 0; i < 8; ++i) { bfrag[i] = blo[i]; bfrag[i + 8] = bhi[i]; }
            acc = __builtin_amdgcn_wmma_f32_16x16x32_bf16(
                false, afrag, false, bfrag, (short)0, acc, false, false);
        }

        float hnew[8];
        if (lane < 16) {
            int col = colsBase + lane;
#pragma unroll
            for (int v = 0; v < 8; ++v) {
                float u = proj[((size_t)v * SEQ + s) * 1024 + col];
                hnew[v] = tanhf(acc[v] + u);
                if (s + 1 < SEQ)  // prefetch next step's u row into cache
                    __builtin_prefetch(&proj[((size_t)v * SEQ + s + 1) * 1024 + col], 0, 3);
            }
        }
        __syncthreads();
        if (lane < 16) {
            int col  = colsBase + lane;
            int lcol = wave * 16 + lane;
#pragma unroll
            for (int v = 0; v < 8; ++v) {
                hbuf[v * HD + lcol] = hnew[v];
                states_bf[((size_t)v * SEQ + s) * 1024 + col] = (__bf16)hnew[v];
                if (s == SEQ - 1) out_state[(size_t)v * 1024 + col] = hnew[v];
            }
        }
        __syncthreads();

        if (tid < BATCH) {
            float sum = 0.0f;
            for (int e = 0; e < HD; ++e) sum += hbuf[tid * HD + e] * cw[e];
            comp[tid] = sum;
        }
        __syncthreads();

        const float tinv = 1.0f / (float)(s + 1);
        const int nxt = cur ^ 1;
        for (int idx = tid; idx < BATCH * HD; idx += 128) {
            int b = idx >> 6;
            int d = idx & 63;
            const float* crow = &cbuf[cur][b * HD];
            float sum = 0.0f;
            for (int e = 0; e < HD; ++e) sum += Alds[d * HD + e] * crow[e];
            cbuf[nxt][idx] = crow[d] - tinv * sum + bv[d] * tinv * comp[b];
        }
        cur = nxt;
        __syncthreads();
    }
}

// ---------------------------------------------------------------------------
extern "C" void kernel_launch(void* const* d_in, const int* in_sizes, int n_in,
                              void* d_out, int out_size, void* d_ws, size_t ws_size,
                              hipStream_t stream) {
    (void)in_sizes; (void)n_in; (void)out_size; (void)ws_size;
    const float* x     = (const float*)d_in[0];
    const float* W_in  = (const float*)d_in[1];
    const float* b_in  = (const float*)d_in[2];
    const float* sw    = (const float*)d_in[3];
    const float* hw    = (const float*)d_in[4];
    const float* cwp   = (const float*)d_in[5];
    const float* Ap    = (const float*)d_in[6];
    const float* Bv    = (const float*)d_in[7];
    const float* W_out = (const float*)d_in[8];
    float* out = (float*)d_out;

    const int M = BATCH * SEQ;   // 16384
    const int N = 1024;
    const int K = 1024;

    // Workspace layout (all sizes 4KB-multiples):
    //   proj      : M*N fp32   = 64 MB
    //   xbf       : M*K bf16   = 32 MB
    //   states_bf : M*N bf16   = 32 MB
    //   WinT      : N*K bf16   =  2 MB
    //   WoutT     : N*K bf16   =  2 MB
    float*  proj      = (float*)d_ws;
    __bf16* xbf       = (__bf16*)(proj + (size_t)M * N);
    __bf16* states_bf = xbf + (size_t)M * K;
    __bf16* WinT      = states_bf + (size_t)M * N;
    __bf16* WoutT     = WinT + (size_t)N * K;

    // One-time precision/layout conversion passes.
    cvt_f32_bf16<<<(M * K / 4 + 255) / 256, 256, 0, stream>>>(x, xbf, M * K / 4);
    dim3 tb(32, 8), tg(N / 32, K / 32);
    transpose_cvt<<<tg, tb, 0, stream>>>(W_in,  WinT,  K, N);
    transpose_cvt<<<tg, tb, 0, stream>>>(W_out, WoutT, K, N);

    dim3 gg(N / BN, M / BM);   // (16, 128)
    gemm_wmma_async<<<gg, 256, 0, stream>>>(xbf, WinT, b_in, proj, M, N, K);
    hippo_scan<<<HEADS, 128, 0, stream>>>(proj, sw, hw, cwp, Ap, Bv,
                                          states_bf, out + (size_t)M * N);
    gemm_wmma_async<<<gg, 256, 0, stream>>>(states_bf, WoutT, nullptr, out, M, N, K);
}
